// PixelwiseSpectralMambaAutoencoder_3753801417447
// MI455X (gfx1250) — compile-verified, matrix operand order checked by reading, not run on hardware
//
#include <hip/hip_runtime.h>
#include <hip/hip_bf16.h>

#define N_PIX 16384
#define C_INC 202
#define TT    51
#define LAYERS 4

typedef __attribute__((ext_vector_type(16))) _Float16 v16h;
typedef __attribute__((ext_vector_type(8)))  _Float16 v8h;
typedef __attribute__((ext_vector_type(8)))  float    v8f;

__device__ __forceinline__ float sigf(float x)  { return 1.f / (1.f + __expf(-x)); }
__device__ __forceinline__ float siluf(float x) { return x * sigf(x); }
__device__ __forceinline__ float geluf(float x) { return 0.5f * x * (1.f + erff(x * 0.70710678118654752f)); }

// ---------------------------------------------------------------------------
// Weight pre-pack: row-major f32 [K][N] -> WMMA-B fragment order, f16.
// For a 16x16x32 f16 WMMA B operand (K=32 rows, N=16 cols per tile):
//   lanes 0-15 : halves 0..7 = K 0..7,   halves 8..15 = K 16..23
//   lanes 16-31: halves 0..7 = K 8..15,  halves 8..15 = K 24..31
//   N = lane & 15
// Packed index: (((kstep*ntiles + ntile)*32 + lane) << 4) + half  -> each lane's
// 16 halves are contiguous (one clause of two global_load_b128 in the GEMM).
// ---------------------------------------------------------------------------
__global__ void pack_b16(const float* __restrict__ src, _Float16* __restrict__ dst,
                         int K, int N, int Npad) {
    int ntiles = Npad >> 4;
    int total = K * Npad;
    for (int i = blockIdx.x * blockDim.x + threadIdx.x; i < total; i += gridDim.x * blockDim.x) {
        int k = i / Npad, n = i % Npad;
        float v = (n < N) ? src[k * N + n] : 0.f;
        int kstep = k >> 5, kin = k & 31;
        int grp  = (kin >> 3) & 1;
        int half = (kin & 7) | ((kin >> 4) << 3);
        int lane = (n & 15) + (grp << 4);
        int nt   = n >> 4;
        dst[(((kstep * ntiles + nt) * 32 + lane) << 4) + half] = (_Float16)v;
    }
}

// One 16x16 output tile: A from LDS f16 (row-major, stride lda), B pre-packed.
// Caller maps mt = tile&3 so mt is wave-invariant and A fragments stay in VGPRs.
__device__ __forceinline__ v8f wmma_tile(const _Float16* A, int lda,
                                         const _Float16* __restrict__ Bp,
                                         int ntiles, int ksteps, int mt, int nt, int lane) {
    v8f acc = {};
    const int arow = mt * 16 + (lane & 15);
    const int koff = (lane >> 4) << 3;
    for (int ks = 0; ks < ksteps; ++ks) {
        const _Float16* ap = A + arow * lda + ks * 32 + koff;
        v8h alo = *(const v8h*)(ap);
        v8h ahi = *(const v8h*)(ap + 16);
        v16h av;
#pragma unroll
        for (int i = 0; i < 8; ++i) { av[i] = alo[i]; av[i + 8] = ahi[i]; }
        v16h bv = *(const v16h*)(Bp + (((ks * ntiles + nt) * 32 + lane) << 4));
        acc = __builtin_amdgcn_wmma_f32_16x16x32_f16(false, av, false, bv,
                                                     (short)0, acc, false, false);
    }
    return acc;
}

struct KParams {
    const float *x, *W_emb, *b_emb, *pos;
    const float *ln_m_g, *ln_m_b, *conv_w, *conv_b, *W_dt, *b_dt, *A_log, *Dp;
    const float *ln_f_g, *ln_f_b, *b1, *b2;
    const float *ln_fin_g, *ln_fin_b, *pool_w;
    const float *enc_ln_g, *enc_ln_b, *enc_W1, *enc_b1, *enc_W2, *enc_b2;
    const float *dec_ln_g, *dec_ln_b, *dec_W1, *dec_b1, *dec_W2, *dec_b2, *dec_W3, *dec_b3;
    const float *ref_w1, *ref_b1, *ref_w2, *ref_b2;
    const _Float16 *win_p, *wx_p, *wout_p, *w1_p, *w2_p;
    float *o_xhat, *o_z, *o_zhat, *o_lik, *o_alpha;
};

// LayerNorm over rows of width 64 (biased var, eps 1e-5) -> f16 dst (WMMA A buffer)
__device__ void ln_rows_h(const float* src, const float* g, const float* b,
                          _Float16* dstH, int tid) {
    for (int r = tid; r < TT; r += 256) {
        const float* row = src + r * 64;
        float m = 0.f;
        for (int j = 0; j < 64; ++j) m += row[j];
        m *= (1.f / 64.f);
        float v = 0.f;
        for (int j = 0; j < 64; ++j) { float d0 = row[j] - m; v += d0 * d0; }
        v *= (1.f / 64.f);
        float inv = rsqrtf(v + 1e-5f);
        for (int j = 0; j < 64; ++j)
            dstH[r * 64 + j] = (_Float16)((row[j] - m) * inv * g[j] + b[j]);
    }
}
__device__ void ln_rows_f(const float* src, const float* g, const float* b,
                          float* dstF, int tid) {
    for (int r = tid; r < TT; r += 256) {
        const float* row = src + r * 64;
        float m = 0.f;
        for (int j = 0; j < 64; ++j) m += row[j];
        m *= (1.f / 64.f);
        float v = 0.f;
        for (int j = 0; j < 64; ++j) { float d0 = row[j] - m; v += d0 * d0; }
        v *= (1.f / 64.f);
        float inv = rsqrtf(v + 1e-5f);
        for (int j = 0; j < 64; ++j)
            dstF[r * 64 + j] = (row[j] - m) * inv * g[j] + b[j];
    }
}

__global__ __launch_bounds__(256)
void msae_main(KParams p) {
    const int tid  = threadIdx.x;
    const int wave = tid >> 5;
    const int lane = tid & 31;
    const int pix  = blockIdx.x;

    __shared__ float sm_h[64 * 64];          // residual stream h (f32); rows 51..63 = dead pad
    __shared__ float sm_xs[64 * 128];        // silu(conv) activations (f32); tail: c1 scratch
    __shared__ float sm_dty[64 * 128];       // dt -> y (in-place); tail: final-LN f32
    __shared__ float sm_dbc[64 * 48];        // dt_pre|B|C (N padded 36->48)
    __shared__ __align__(16) _Float16 sm_a16[64 * 256]; // xz f16 (xc|z) / FF hidden f16
    __shared__ __align__(16) _Float16 sm_u16[64 * 64];  // LN output, WMMA A operand
    __shared__ __align__(16) _Float16 sm_s16[64 * 128]; // xs16 / yg16, WMMA A operand
    __shared__ float sm_red[64];
    __shared__ float sm_tail[1280];

    float* POOLED = sm_tail;
    float* E      = sm_tail + 64;
    float* E1     = sm_tail + 128;
    float* ZH     = sm_tail + 400;
    float* D0     = sm_tail + 416;
    float* D1     = sm_tail + 432;
    float* D2     = sm_tail + 688;
    float* XH     = sm_tail + 752;

    // ---- init (zero-pad rows 51..63 for WMMA A operands; h pad rows are dead)
    for (int i = tid; i < 64 * 64; i += 256) { sm_h[i] = 0.f; sm_u16[i] = (_Float16)0.f; }
    __syncthreads();

    // ---- embedding: tok(51x4) @ W_emb(4x64) + b_emb + pos
    for (int i = tid; i < TT * 64; i += 256) {
        int t = i >> 6, d = i & 63;
        float acc = p.b_emb[d] + p.pos[t * 64 + d];
#pragma unroll
        for (int g = 0; g < 4; ++g) {
            int ci = t * 4 + g;
            float xv = (ci < C_INC) ? p.x[(size_t)pix * C_INC + ci] : 0.f;
            acc += xv * p.W_emb[g * 64 + d];
        }
        sm_h[t * 64 + d] = acc;
    }
    __syncthreads();

    for (int l = 0; l < LAYERS; ++l) {
        // ---- u = LN(h); shared by both directions (flip folded into indexing later)
        ln_rows_h(sm_h, p.ln_m_g + l * 64, p.ln_m_b + l * 64, sm_u16, tid);
        __syncthreads();

        for (int dir = 0; dir < 2; ++dir) {
            const int ld = l * 2 + dir;

            // ---- GEMM1: xz = u @ W_in (M=64 K=64 N=256); flip(u)@W == flip(u@W).
            // Unconditional f16 store into unified xz buffer (no per-lane branch).
            const _Float16* bp = p.win_p + (size_t)ld * 16384;
            __builtin_prefetch(bp, 0, 1);
            for (int tile = wave; tile < 64; tile += 8) {
                int mt = tile & 3, nt = tile >> 2;
                v8f c = wmma_tile(sm_u16, 64, bp, 16, 2, mt, nt, lane);
                int col = nt * 16 + (lane & 15);
#pragma unroll
                for (int r = 0; r < 8; ++r) {
                    int row = mt * 16 + r + ((lane >> 4) << 3);
                    sm_a16[row * 256 + col] = (_Float16)c[r];
                }
            }
            __syncthreads();

            // ---- causal depthwise conv (K=4) + SiLU -> xs (f32) and xs16 (A operand)
            // mamba time tau; physical row pr = dir ? 50-tau : tau. xc = sm_a16[:, :128]
            for (int i = tid; i < 64 * 128; i += 256) {
                int t = i >> 7, c = i & 127;
                float v = 0.f;
                if (t < TT) {
                    float acc = p.conv_b[ld * 128 + c];
#pragma unroll
                    for (int k = 0; k < 4; ++k) {
                        int ts = t + k - 3;
                        if (ts >= 0) {
                            int pr = dir ? (TT - 1 - ts) : ts;
                            acc += (float)sm_a16[pr * 256 + c] * p.conv_w[(ld * 128 + c) * 4 + k];
                        }
                    }
                    v = siluf(acc);
                }
                sm_xs[i]  = v;
                sm_s16[i] = (_Float16)v;
            }
            __syncthreads();

            // ---- GEMM2: dbc = xs @ W_x (M=64 K=128 N=36->48)
            const _Float16* bpx = p.wx_p + (size_t)ld * 6144;
            __builtin_prefetch(bpx, 0, 1);
            for (int tile = wave; tile < 12; tile += 8) {
                int mt = tile & 3, nt = tile >> 2;
                v8f c = wmma_tile(sm_s16, 128, bpx, 3, 4, mt, nt, lane);
                int col = nt * 16 + (lane & 15);
#pragma unroll
                for (int r = 0; r < 8; ++r) {
                    int row = mt * 16 + r + ((lane >> 4) << 3);
                    sm_dbc[row * 48 + col] = c[r];
                }
            }
            __syncthreads();

            // ---- dt = softplus(dbc[:, :4] @ W_dt + b_dt) -> sm_dty
            for (int i = tid; i < TT * 128; i += 256) {
                int t = i >> 7, c = i & 127;
                float acc = p.b_dt[ld * 128 + c];
#pragma unroll
                for (int r = 0; r < 4; ++r)
                    acc += sm_dbc[t * 48 + r] * p.W_dt[(ld * 4 + r) * 128 + c];
                sm_dty[t * 128 + c] = (acc > 20.f) ? acc : log1pf(__expf(acc));
            }
            __syncthreads();

            // ---- selective scan: thread = channel, 16 states in registers
            if (tid < 128) {
                int c = tid;
                float As[16], hst[16];
#pragma unroll
                for (int s = 0; s < 16; ++s) {
                    As[s] = -__expf(p.A_log[((size_t)ld * 128 + c) * 16 + s]);
                    hst[s] = 0.f;
                }
                float Dpv = p.Dp[ld * 128 + c];
                for (int t = 0; t < TT; ++t) {
                    float dtv = sm_dty[t * 128 + c];
                    float xv  = sm_xs[t * 128 + c];
                    float dx  = dtv * xv;
                    float acc = 0.f;
#pragma unroll
                    for (int s = 0; s < 16; ++s) {
                        hst[s] = __expf(dtv * As[s]) * hst[s] + dx * sm_dbc[t * 48 + 4 + s];
                        acc += hst[s] * sm_dbc[t * 48 + 20 + s];
                    }
                    sm_dty[t * 128 + c] = acc + xv * Dpv;   // y overwrites dt in place
                }
            }
            __syncthreads();

            // ---- gate: yg = y * silu(z) -> A operand (f16); z = sm_a16[:, 128:]
            for (int i = tid; i < 64 * 128; i += 256) {
                int t = i >> 7, c = i & 127;
                float v = 0.f;
                if (t < TT) {
                    int pr = dir ? (TT - 1 - t) : t;      // z lives in physical time
                    float zg = (float)sm_a16[pr * 256 + 128 + c];
                    v = sm_dty[i] * siluf(zg);
                }
                sm_s16[i] = (_Float16)v;
            }
            __syncthreads();

            // ---- GEMM3: out = yg @ W_out (M=64 K=128 N=64), accumulate into h.
            // Branch-free: out-of-range rows redirect into dead pad rows of sm_h.
            const _Float16* bpo = p.wout_p + (size_t)ld * 8192;
            __builtin_prefetch(bpo, 0, 1);
            for (int tile = wave; tile < 16; tile += 8) {
                int mt = tile & 3, nt = tile >> 2;
                v8f c = wmma_tile(sm_s16, 128, bpo, 4, 4, mt, nt, lane);
                int col = nt * 16 + (lane & 15);
#pragma unroll
                for (int r = 0; r < 8; ++r) {
                    int row = mt * 16 + r + ((lane >> 4) << 3);
                    int tr  = dir ? (TT - 1 - row) : row;   // un-flip
                    int dstrow = (row < TT) ? tr : row;     // select, not branch
                    sm_h[dstrow * 64 + col] += c[r];
                }
            }
            __syncthreads();
        } // dir

        // ---- FFN: LN -> W1(64x256)+gelu -> W2(256x64) residual
        ln_rows_h(sm_h, p.ln_f_g + l * 64, p.ln_f_b + l * 64, sm_u16, tid);
        __syncthreads();
        const _Float16* bp1 = p.w1_p + (size_t)l * 16384;
        __builtin_prefetch(bp1, 0, 1);
        for (int tile = wave; tile < 64; tile += 8) {
            int mt = tile & 3, nt = tile >> 2;
            v8f c = wmma_tile(sm_u16, 64, bp1, 16, 2, mt, nt, lane);
            int col = nt * 16 + (lane & 15);
#pragma unroll
            for (int r = 0; r < 8; ++r) {
                int row = mt * 16 + r + ((lane >> 4) << 3);
                sm_a16[row * 256 + col] = (_Float16)geluf(c[r] + p.b1[l * 256 + col]);
            }
        }
        __syncthreads();
        const _Float16* bp2 = p.w2_p + (size_t)l * 16384;
        __builtin_prefetch(bp2, 0, 1);
        for (int tile = wave; tile < 16; tile += 8) {
            int mt = tile & 3, nt = tile >> 2;
            v8f c = wmma_tile(sm_a16, 256, bp2, 4, 8, mt, nt, lane);
            int col = nt * 16 + (lane & 15);
#pragma unroll
            for (int r = 0; r < 8; ++r) {
                int row = mt * 16 + r + ((lane >> 4) << 3);
                sm_h[row * 64 + col] += c[r] + p.b2[l * 64 + col];   // pad rows absorb junk
            }
        }
        __syncthreads();
    } // layers

    // ---- final LN (f32 into sm_dty), attention pool
    ln_rows_f(sm_h, p.ln_fin_g, p.ln_fin_b, sm_dty, tid);
    __syncthreads();
    if (tid < TT) {
        float acc = 0.f;
        for (int d = 0; d < 64; ++d) acc += sm_dty[tid * 64 + d] * p.pool_w[d];
        sm_red[tid] = acc;
    }
    __syncthreads();
    if (tid == 0) {
        float mx = -1e30f;
        for (int t = 0; t < TT; ++t) mx = fmaxf(mx, sm_red[t]);
        float s = 0.f;
        for (int t = 0; t < TT; ++t) { float e = __expf(sm_red[t] - mx); sm_red[t] = e; s += e; }
        float inv = 1.f / s;
        for (int t = 0; t < TT; ++t) sm_red[t] *= inv;
    }
    __syncthreads();
    if (tid < TT) p.o_alpha[(size_t)pix * TT + tid] = sm_red[tid];
    if (tid < 64) {
        float acc = 0.f;
        for (int t = 0; t < TT; ++t) acc += sm_red[t] * sm_dty[t * 64 + tid];
        POOLED[tid] = acc;
    }
    __syncthreads();

    // ---- encoder
    if (tid == 0) {
        float m = 0.f;
        for (int d = 0; d < 64; ++d) m += POOLED[d];
        m *= (1.f / 64.f);
        float v = 0.f;
        for (int d = 0; d < 64; ++d) { float dd = POOLED[d] - m; v += dd * dd; }
        v *= (1.f / 64.f);
        sm_red[62] = m; sm_red[63] = rsqrtf(v + 1e-5f);
    }
    __syncthreads();
    if (tid < 64)
        E[tid] = (POOLED[tid] - sm_red[62]) * sm_red[63] * p.enc_ln_g[tid] + p.enc_ln_b[tid];
    __syncthreads();
    {
        float acc = p.enc_b1[tid];
        for (int d = 0; d < 64; ++d) acc += E[d] * p.enc_W1[d * 256 + tid];
        E1[tid] = geluf(acc);
    }
    __syncthreads();
    if (tid < 16) {
        float acc = p.enc_b2[tid];
        for (int j = 0; j < 256; ++j) acc += E1[j] * p.enc_W2[j * 16 + tid];
        float zh = rintf(acc);             // straight-through round (RNE, like jnp.round)
        ZH[tid] = zh;
        p.o_z[(size_t)pix * 16 + tid]    = acc;
        p.o_zhat[(size_t)pix * 16 + tid] = zh;
        p.o_lik[(size_t)pix * 16 + tid]  = sigf(zh + 0.5f) - sigf(zh - 0.5f);
    }
    __syncthreads();

    // ---- decoder
    if (tid == 0) {
        float m = 0.f;
        for (int i = 0; i < 16; ++i) m += ZH[i];
        m *= (1.f / 16.f);
        float v = 0.f;
        for (int i = 0; i < 16; ++i) { float dd = ZH[i] - m; v += dd * dd; }
        v *= (1.f / 16.f);
        float inv = rsqrtf(v + 1e-5f);
        for (int i = 0; i < 16; ++i)
            D0[i] = (ZH[i] - m) * inv * p.dec_ln_g[i] + p.dec_ln_b[i];
    }
    __syncthreads();
    {
        float acc = p.dec_b1[tid];
        for (int i = 0; i < 16; ++i) acc += D0[i] * p.dec_W1[i * 256 + tid];
        D1[tid] = geluf(acc);
    }
    __syncthreads();
    if (tid < 64) {
        float acc = p.dec_b2[tid];
        for (int j = 0; j < 256; ++j) acc += D1[j] * p.dec_W2[j * 64 + tid];
        D2[tid] = geluf(acc);
    }
    __syncthreads();
    if (tid < C_INC) {
        float acc = p.dec_b3[tid];
        for (int k = 0; k < 64; ++k) acc += D2[k] * p.dec_W3[k * C_INC + tid];
        XH[tid] = acc;
    }
    __syncthreads();

    // ---- refinement conv (1->16->1, k=3, pad 1), c1 staged in sm_xs
    for (int i = tid; i < 16 * C_INC; i += 256) {
        int o = i / C_INC, c = i % C_INC;
        float acc = p.ref_b1[o];
#pragma unroll
        for (int k = 0; k < 3; ++k) {
            int cc = c + k - 1;
            if (cc >= 0 && cc < C_INC) acc += XH[cc] * p.ref_w1[o * 3 + k];
        }
        sm_xs[o * 204 + c] = acc;
    }
    __syncthreads();
    if (tid < C_INC) {
        int c = tid;
        float acc = p.ref_b2[0];
        for (int o = 0; o < 16; ++o) {
#pragma unroll
            for (int k = 0; k < 3; ++k) {
                int cc = c + k - 1;
                if (cc >= 0 && cc < C_INC)
                    acc += geluf(sm_xs[o * 204 + cc]) * p.ref_w2[o * 3 + k];
            }
        }
        p.o_xhat[(size_t)pix * C_INC + c] = sigf(XH[c] + acc);
    }
}

extern "C" void kernel_launch(void* const* d_in, const int* in_sizes, int n_in,
                              void* d_out, int out_size, void* d_ws, size_t ws_size,
                              hipStream_t stream) {
    (void)in_sizes; (void)n_in; (void)out_size; (void)ws_size;

    // setup_inputs() flat order
    const float* x       = (const float*)d_in[0];
    const float* W_emb   = (const float*)d_in[1];
    const float* b_emb   = (const float*)d_in[2];
    const float* pos     = (const float*)d_in[3];
    const float* ln_m_g  = (const float*)d_in[4];
    const float* ln_m_b  = (const float*)d_in[5];
    const float* W_in    = (const float*)d_in[6];
    const float* conv_w  = (const float*)d_in[7];
    const float* conv_b  = (const float*)d_in[8];
    const float* W_x     = (const float*)d_in[9];
    const float* W_dt    = (const float*)d_in[10];
    const float* b_dt    = (const float*)d_in[11];
    const float* A_log   = (const float*)d_in[12];
    const float* Dp      = (const float*)d_in[13];
    const float* W_out   = (const float*)d_in[14];
    const float* ln_f_g  = (const float*)d_in[15];
    const float* ln_f_b  = (const float*)d_in[16];
    const float* W1      = (const float*)d_in[17];
    const float* b1      = (const float*)d_in[18];
    const float* W2      = (const float*)d_in[19];
    const float* b2      = (const float*)d_in[20];
    const float* ln_fin_g = (const float*)d_in[21];
    const float* ln_fin_b = (const float*)d_in[22];
    const float* pool_w   = (const float*)d_in[23];
    const float* enc_ln_g = (const float*)d_in[24];
    const float* enc_ln_b = (const float*)d_in[25];
    const float* enc_W1   = (const float*)d_in[26];
    const float* enc_b1   = (const float*)d_in[27];
    const float* enc_W2   = (const float*)d_in[28];
    const float* enc_b2   = (const float*)d_in[29];
    const float* dec_ln_g = (const float*)d_in[30];
    const float* dec_ln_b = (const float*)d_in[31];
    const float* dec_W1   = (const float*)d_in[32];
    const float* dec_b1   = (const float*)d_in[33];
    const float* dec_W2   = (const float*)d_in[34];
    const float* dec_b2   = (const float*)d_in[35];
    const float* dec_W3   = (const float*)d_in[36];
    const float* dec_b3   = (const float*)d_in[37];
    const float* ref_w1   = (const float*)d_in[38];
    const float* ref_b1   = (const float*)d_in[39];
    const float* ref_w2   = (const float*)d_in[40];
    const float* ref_b2   = (const float*)d_in[41];

    // Workspace: pre-packed f16 WMMA-B weights (~736 KB)
    _Float16* ws16   = (_Float16*)d_ws;
    _Float16* win_p  = ws16;                       // 8 * (2*16*32*16)  = 8*16384
    _Float16* wx_p   = win_p  + 8 * 16384;         // 8 * (4*3*32*16)   = 8*6144
    _Float16* wout_p = wx_p   + 8 * 6144;          // 8 * (4*4*32*16)   = 8*8192
    _Float16* w1_p   = wout_p + 8 * 8192;          // 4 * 16384
    _Float16* w2_p   = w1_p   + 4 * 16384;         // 4 * 16384

    for (int ld = 0; ld < 8; ++ld) {
        pack_b16<<<16, 256, 0, stream>>>(W_in  + (size_t)ld * 64 * 256, win_p  + (size_t)ld * 16384, 64, 256, 256);
        pack_b16<<<8,  256, 0, stream>>>(W_x   + (size_t)ld * 128 * 36, wx_p   + (size_t)ld * 6144, 128, 36, 48);
        pack_b16<<<8,  256, 0, stream>>>(W_out + (size_t)ld * 128 * 64, wout_p + (size_t)ld * 8192, 128, 64, 64);
    }
    for (int l = 0; l < 4; ++l) {
        pack_b16<<<16, 256, 0, stream>>>(W1 + (size_t)l * 64 * 256, w1_p + (size_t)l * 16384, 64, 256, 256);
        pack_b16<<<16, 256, 0, stream>>>(W2 + (size_t)l * 256 * 64, w2_p + (size_t)l * 16384, 256, 64, 64);
    }

    KParams kp;
    kp.x = x; kp.W_emb = W_emb; kp.b_emb = b_emb; kp.pos = pos;
    kp.ln_m_g = ln_m_g; kp.ln_m_b = ln_m_b; kp.conv_w = conv_w; kp.conv_b = conv_b;
    kp.W_dt = W_dt; kp.b_dt = b_dt; kp.A_log = A_log; kp.Dp = Dp;
    kp.ln_f_g = ln_f_g; kp.ln_f_b = ln_f_b; kp.b1 = b1; kp.b2 = b2;
    kp.ln_fin_g = ln_fin_g; kp.ln_fin_b = ln_fin_b; kp.pool_w = pool_w;
    kp.enc_ln_g = enc_ln_g; kp.enc_ln_b = enc_ln_b;
    kp.enc_W1 = enc_W1; kp.enc_b1 = enc_b1; kp.enc_W2 = enc_W2; kp.enc_b2 = enc_b2;
    kp.dec_ln_g = dec_ln_g; kp.dec_ln_b = dec_ln_b;
    kp.dec_W1 = dec_W1; kp.dec_b1 = dec_b1; kp.dec_W2 = dec_W2; kp.dec_b2 = dec_b2;
    kp.dec_W3 = dec_W3; kp.dec_b3 = dec_b3;
    kp.ref_w1 = ref_w1; kp.ref_b1 = ref_b1; kp.ref_w2 = ref_w2; kp.ref_b2 = ref_b2;
    kp.win_p = win_p; kp.wx_p = wx_p; kp.wout_p = wout_p; kp.w1_p = w1_p; kp.w2_p = w2_p;

    float* out = (float*)d_out;
    kp.o_xhat  = out;
    kp.o_z     = out + (size_t)N_PIX * C_INC;
    kp.o_zhat  = kp.o_z    + (size_t)N_PIX * 16;
    kp.o_lik   = kp.o_zhat + (size_t)N_PIX * 16;
    kp.o_alpha = kp.o_lik  + (size_t)N_PIX * 16;

    msae_main<<<N_PIX, 256, 0, stream>>>(kp);
}
